// TokenMoE_77360950935847
// MI455X (gfx1250) — matrix operand order repaired and checked by reading, not compiled, not run on hardware
//
#include <hip/hip_runtime.h>
#include <hip/hip_bf16.h>
#include <math.h>

// ---------------- problem constants ----------------
#define DMODEL 1024
#define NEXP   8
#define DFF    4096
#define NTOK   8192          // B*T = 4*2048
#define BM     64            // routed rows per workgroup
#define BF     64            // F-chunk
#define MAXSLOTS (NTOK*2 + NEXP*BM)   // 16896 (top-2 slots + per-expert pad)
#define NTILES  (MAXSLOTS/BM)         // 264

typedef __attribute__((ext_vector_type(16))) __bf16 v16bf;
typedef __attribute__((ext_vector_type(8)))  __bf16 v8bf;
typedef __attribute__((ext_vector_type(8)))  float  v8f;

union Frag { v16bf v; v8bf h[2]; };

// ---------------- workspace layout (bytes) ----------------
static constexpr size_t W1B_OFF  = 0;
static constexpr size_t W1B_SZ   = (size_t)NEXP*DFF*DMODEL*2;   // 64 MB bf16
static constexpr size_t W2B_OFF  = W1B_OFF + W1B_SZ;
static constexpr size_t W2B_SZ   = (size_t)NEXP*DMODEL*DFF*2;   // 64 MB bf16
static constexpr size_t YBUF_OFF = W2B_OFF + W2B_SZ;
static constexpr size_t YBUF_SZ  = (size_t)NTOK*2*DMODEL*4;     // 64 MB f32
static constexpr size_t TE_OFF   = YBUF_OFF + YBUF_SZ;          // int [NTOK*2]
static constexpr size_t TW_OFF   = TE_OFF + (size_t)NTOK*2*4;   // f32 [NTOK*2]
static constexpr size_t RT_OFF   = TW_OFF + (size_t)NTOK*2*4;   // int [MAXSLOTS] token id
static constexpr size_t RK_OFF   = RT_OFF + (size_t)MAXSLOTS*4; // int [MAXSLOTS] which-k
static constexpr size_t RW_OFF   = RK_OFF + (size_t)MAXSLOTS*4; // f32 [MAXSLOTS] routing weight
static constexpr size_t CNT_OFF  = RW_OFF + (size_t)MAXSLOTS*4; // int [NEXP]
static constexpr size_t OFFS_OFF = CNT_OFF + 8*4;               // int [NEXP+1]
static constexpr size_t CUR_OFF  = OFFS_OFF + 9*4;              // int [NEXP]

// ---------------- LDS layout for k_ffn ----------------
#define XS_STRIDE 1040   // 1024 + 16 pad (bf16 elems), keeps 16B alignment
#define HS_STRIDE 72
static constexpr size_t SM_XS   = 0;                                  // bf16[BM][XS_STRIDE]
static constexpr size_t SM_W1S  = SM_XS  + (size_t)BM*XS_STRIDE*2;    // bf16[BF][XS_STRIDE]
static constexpr size_t SM_HS   = SM_W1S + (size_t)BF*XS_STRIDE*2;    // bf16[BM][HS_STRIDE]
static constexpr size_t SM_TK   = SM_HS  + (size_t)BM*HS_STRIDE*2;    // int[BM]
static constexpr size_t SM_DST  = SM_TK  + BM*4;                      // int[BM]
static constexpr size_t SM_RW   = SM_DST + BM*4;                      // f32[BM]
static constexpr size_t SM_FLAG = SM_RW  + BM*4;                      // int
static constexpr size_t SMEM_BYTES = ((SM_FLAG + 4 + 255) / 256) * 256; // ~276 KB < 320 KB

__device__ __forceinline__ unsigned short f2bfu(float f) {
  unsigned u = __builtin_bit_cast(unsigned, f);
  unsigned r = (u + 0x7FFFu + ((u >> 16) & 1u)) >> 16;     // round-to-nearest-even
  return (unsigned short)r;
}
__device__ __forceinline__ __bf16 f2bf(float f) {
  return __builtin_bit_cast(__bf16, f2bfu(f));
}
__device__ __forceinline__ float gelu_erf(float h) {
  return 0.5f * h * (1.0f + erff(h * 0.70710678118654752f));
}
__device__ __forceinline__ void zero8(v8f& v) {
#pragma unroll
  for (int i = 0; i < 8; ++i) v[i] = 0.0f;
}

// A-fragment (16x32 bf16): lane = M (lane&15); K interleave per ISA table:
// elements 0..7 -> K = kh*8 + i, elements 8..15 -> K = 16 + kh*8 + (i-8), kh = lane>>4.
template <typename PT>
__device__ __forceinline__ Frag load_frag_a(const PT* base, int row0, int stride,
                                            int kbase, int lane) {
  Frag f;
  int m = lane & 15, kh = (lane >> 4) & 1;
  const __bf16* p = (const __bf16*)base + (size_t)(row0 + m) * stride + kbase;
  f.h[0] = *(const v8bf*)(p + kh * 8);
  f.h[1] = *(const v8bf*)(p + 16 + kh * 8);
  return f;
}
// B-fragment (32x16 bf16): lane = N (lane&15); K interleave per ISA table (same as A).
template <typename PT>
__device__ __forceinline__ Frag load_frag_b(const PT* base, int col0, long stride,
                                            int kbase, int lane) {
  Frag f;
  int n = lane & 15, kh = (lane >> 4) & 1;
  const __bf16* p = (const __bf16*)base + (size_t)(col0 + n) * stride + kbase;
  f.h[0] = *(const v8bf*)(p + kh * 8);
  f.h[1] = *(const v8bf*)(p + 16 + kh * 8);
  return f;
}

// ================= kernels =================

__global__ __launch_bounds__(32) void k_init(int* cnt) {
  if (threadIdx.x < NEXP) cnt[threadIdx.x] = 0;
}

__global__ __launch_bounds__(256) void k_cvt(const float* __restrict__ src,
                                             __bf16* __restrict__ dst, int n4) {
  int i = blockIdx.x * 256 + threadIdx.x;
  if (i >= n4) return;
  float4 v = ((const float4*)src)[i];
  unsigned p0 = (unsigned)f2bfu(v.x) | ((unsigned)f2bfu(v.y) << 16);
  unsigned p1 = (unsigned)f2bfu(v.z) | ((unsigned)f2bfu(v.w) << 16);
  ((uint2*)dst)[i] = make_uint2(p0, p1);
}

__global__ __launch_bounds__(256) void k_gate(const float* __restrict__ x,
                                              const float* __restrict__ gw,
                                              int* __restrict__ te, float* __restrict__ tw,
                                              int* __restrict__ cnt) {
  __shared__ float gws[NEXP * DMODEL];   // 32 KB
  int tid = threadIdx.x;
  for (int i = tid; i < NEXP * DMODEL; i += 256) gws[i] = gw[i];
  __syncthreads();

  int n = blockIdx.x * 256 + tid;
  const float* xr = x + (size_t)n * DMODEL;
  float acc[NEXP];
#pragma unroll
  for (int e = 0; e < NEXP; ++e) acc[e] = 0.0f;
  for (int j = 0; j < DMODEL; j += 4) {
    float4 xv = *(const float4*)(xr + j);
#pragma unroll
    for (int e = 0; e < NEXP; ++e) {
      const float* g = &gws[e * DMODEL + j];
      acc[e] += xv.x * g[0] + xv.y * g[1] + xv.z * g[2] + xv.w * g[3];
    }
  }
  float m = acc[0];
#pragma unroll
  for (int e = 1; e < NEXP; ++e) m = fmaxf(m, acc[e]);
  float p[NEXP], s = 0.0f;
#pragma unroll
  for (int e = 0; e < NEXP; ++e) { p[e] = expf(acc[e] - m); s += p[e]; }
  float inv = 1.0f / s;
#pragma unroll
  for (int e = 0; e < NEXP; ++e) p[e] *= inv;
  // top-2: ties -> lowest index (matches lax.top_k)
  int e1 = 0; float v1 = p[0];
#pragma unroll
  for (int e = 1; e < NEXP; ++e) if (p[e] > v1) { v1 = p[e]; e1 = e; }
  int e2 = (e1 == 0) ? 1 : 0; float v2 = p[e2];
#pragma unroll
  for (int e = 0; e < NEXP; ++e)
    if (e != e1 && p[e] > v2) { v2 = p[e]; e2 = e; }

  te[n * 2 + 0] = e1; tw[n * 2 + 0] = v1;
  te[n * 2 + 1] = e2; tw[n * 2 + 1] = v2;
  atomicAdd(&cnt[e1], 1);
  atomicAdd(&cnt[e2], 1);
}

__global__ __launch_bounds__(256) void k_offs(const int* __restrict__ cnt,
                                              int* __restrict__ offs, int* __restrict__ cur,
                                              int* __restrict__ rowTok) {
  int tid = threadIdx.x;
  if (tid == 0) {
    int run = 0;
#pragma unroll
    for (int e = 0; e < NEXP; ++e) {
      offs[e] = run;
      cur[e] = run;
      run += (cnt[e] + (BM - 1)) & ~(BM - 1);
    }
    offs[NEXP] = run;
  }
  for (int i = tid; i < MAXSLOTS; i += 256) rowTok[i] = -1;
}

__global__ __launch_bounds__(256) void k_assign(const int* __restrict__ te,
                                                const float* __restrict__ tw,
                                                int* __restrict__ cur,
                                                int* __restrict__ rowTok,
                                                int* __restrict__ rowK,
                                                float* __restrict__ rowW) {
  int n = blockIdx.x * 256 + threadIdx.x;
#pragma unroll
  for (int k = 0; k < 2; ++k) {
    int e = te[n * 2 + k];
    int pos = atomicAdd(&cur[e], 1);
    rowTok[pos] = n;
    rowK[pos] = k;
    rowW[pos] = tw[n * 2 + k];
  }
}

// Fused expert FFN over one 64-row routed tile: Y = (gelu(X*W1^T+b1))*W2^T + b2, scaled.
__global__ __launch_bounds__(512) void
k_ffn(const float* __restrict__ x, const __bf16* __restrict__ w1b,
      const __bf16* __restrict__ w2b, const float* __restrict__ b1,
      const float* __restrict__ b2, const int* __restrict__ rowTok,
      const int* __restrict__ rowK, const float* __restrict__ rowW,
      const int* __restrict__ offs, float* __restrict__ ybuf) {
  extern __shared__ __align__(16) unsigned char smem[];
  __bf16* Xs  = (__bf16*)(smem + SM_XS);
  __bf16* W1s = (__bf16*)(smem + SM_W1S);
  __bf16* Hs  = (__bf16*)(smem + SM_HS);
  int*    tkS  = (int*)(smem + SM_TK);
  int*    dstS = (int*)(smem + SM_DST);
  float*  rwS  = (float*)(smem + SM_RW);
  int*    flagS = (int*)(smem + SM_FLAG);

  const int tid = threadIdx.x;
  const int wv = tid >> 5;          // 0..15
  const int lane = tid & 31;
  const int s0 = blockIdx.x * BM;

  if (tid == 0) flagS[0] = 0;
  if (tid < BM) {
    int slot = s0 + tid;
    int tok = rowTok[slot];
    tkS[tid] = tok;
    rwS[tid] = rowW[slot];
    dstS[tid] = (tok >= 0) ? (tok * 2 + rowK[slot]) : -1;
  }
  __syncthreads();
  if (tid < BM && tkS[tid] >= 0) atomicOr(flagS, 1);
  __syncthreads();
  if (!flagS[0]) return;            // fully padded tile

  // expert of this tile (segments are BM-aligned)
  int e = 0;
#pragma unroll
  for (int i = 1; i < NEXP; ++i) if (s0 >= offs[i]) e = i;

  const __bf16* w1e = w1b + (size_t)e * DFF * DMODEL;
  const __bf16* w2e = w2b + (size_t)e * DMODEL * DFF;
  const float*  b1e = b1 + (size_t)e * DFF;
  const float*  b2e = b2 + (size_t)e * DMODEL;

  // gather + convert X tile into LDS (bf16)
  for (int i = tid; i < BM * (DMODEL / 4); i += 512) {
    int r = i >> 8;
    int c = (i & 255) * 4;
    int tok = tkS[r]; if (tok < 0) tok = 0;
    float4 xv = *(const float4*)(x + (size_t)tok * DMODEL + c);
    unsigned p0 = (unsigned)f2bfu(xv.x) | ((unsigned)f2bfu(xv.y) << 16);
    unsigned p1 = (unsigned)f2bfu(xv.z) | ((unsigned)f2bfu(xv.w) << 16);
    *(uint2*)(Xs + (size_t)r * XS_STRIDE + c) = make_uint2(p0, p1);
  }

  // stage-A wave tiling: Hc(64x64) -> 4x4 fragments, one per wave
  const int mi_a = wv & 3, fi_a = wv >> 2;
  // stage-B: wave owns output cols [64*wv, 64*wv+64)
  v8f acc[4][4];
#pragma unroll
  for (int mi = 0; mi < 4; ++mi)
#pragma unroll
    for (int ni = 0; ni < 4; ++ni) zero8(acc[mi][ni]);

  const int nn = lane & 15, khl = (lane >> 4) & 1;

  for (int f0 = 0; f0 < DFF; f0 += BF) {
    __syncthreads();                               // stage-B(prev) done with Hs/W1s
    // stage W1 chunk (rows f0..f0+63) into LDS, bf16
    for (int i = tid; i < BF * (DMODEL / 8); i += 512) {
      int r = i >> 7;
      int c = (i & 127) * 8;
      *(v8bf*)(W1s + (size_t)r * XS_STRIDE + c) =
          *(const v8bf*)(w1e + (size_t)(f0 + r) * DMODEL + c);
    }
    if (f0 + BF < DFF && tid < BF)                 // hint next chunk into L2
      __builtin_prefetch(w1e + (size_t)(f0 + BF + tid) * DMODEL, 0, 1);
    __syncthreads();

    // ---- stage A: Hc = X * W1_chunk^T  (K = 1024) ----
    v8f hacc; zero8(hacc);
#pragma unroll 4
    for (int kb = 0; kb < DMODEL / 32; ++kb) {
      Frag a = load_frag_a(Xs, 16 * mi_a, XS_STRIDE, kb * 32, lane);
      Frag b = load_frag_b(W1s, 16 * fi_a, XS_STRIDE, kb * 32, lane);
      hacc = __builtin_amdgcn_wmma_f32_16x16x32_bf16(false, a.v, false, b.v,
                                                     (short)0, hacc, false, false);
    }
    float bias1 = b1e[f0 + 16 * fi_a + nn];
#pragma unroll
    for (int r = 0; r < 8; ++r) {
      float h = gelu_erf(hacc[r] + bias1);
      Hs[(size_t)(16 * mi_a + r + 8 * khl) * HS_STRIDE + 16 * fi_a + nn] = f2bf(h);
    }
    __syncthreads();

    // ---- stage B: Yacc += Hc * W2_chunk^T (K = 64) ----
#pragma unroll
    for (int kb2 = 0; kb2 < 2; ++kb2) {
      Frag ha[4];
#pragma unroll
      for (int mi = 0; mi < 4; ++mi)
        ha[mi] = load_frag_a(Hs, 16 * mi, HS_STRIDE, kb2 * 32, lane);
#pragma unroll
      for (int ni = 0; ni < 4; ++ni) {
        int c0 = 64 * wv + 16 * ni;
        Frag wb = load_frag_b(w2e, c0, (long)DFF, f0 + kb2 * 32, lane);
#pragma unroll
        for (int mi = 0; mi < 4; ++mi)
          acc[mi][ni] = __builtin_amdgcn_wmma_f32_16x16x32_bf16(
              false, ha[mi].v, false, wb.v, (short)0, acc[mi][ni], false, false);
      }
    }
  }

  // epilogue: +b2, scale by routing weight, scatter to ybuf[(tok*2+k)][D]
#pragma unroll
  for (int ni = 0; ni < 4; ++ni) {
    int col = 64 * wv + 16 * ni + nn;
    float bias2 = b2e[col];
#pragma unroll
    for (int mi = 0; mi < 4; ++mi) {
#pragma unroll
      for (int r = 0; r < 8; ++r) {
        int row = 16 * mi + r + 8 * khl;
        int d = dstS[row];
        if (d >= 0)
          ybuf[(size_t)d * DMODEL + col] = (acc[mi][ni][r] + bias2) * rwS[row];
      }
    }
  }
}

__global__ __launch_bounds__(256) void k_combine(const float* __restrict__ ybuf,
                                                 float* __restrict__ out) {
  int i = blockIdx.x * 256 + threadIdx.x;          // float4 index over [NTOK][D]
  int n = i >> 8;
  int c = (i & 255) * 4;
  float4 a = *(const float4*)(ybuf + ((size_t)n * 2 + 0) * DMODEL + c);
  float4 b = *(const float4*)(ybuf + ((size_t)n * 2 + 1) * DMODEL + c);
  float4 o = make_float4(a.x + b.x, a.y + b.y, a.z + b.z, a.w + b.w);
  *(float4*)(out + (size_t)n * DMODEL + c) = o;
}

// ================= launch =================
extern "C" void kernel_launch(void* const* d_in, const int* in_sizes, int n_in,
                              void* d_out, int out_size, void* d_ws, size_t ws_size,
                              hipStream_t stream) {
  const float* x      = (const float*)d_in[0];
  const float* gate_w = (const float*)d_in[1];
  const float* w1     = (const float*)d_in[2];
  const float* b1     = (const float*)d_in[3];
  const float* w2     = (const float*)d_in[4];
  const float* b2     = (const float*)d_in[5];
  float* out = (float*)d_out;

  char* ws = (char*)d_ws;
  __bf16* w1b   = (__bf16*)(ws + W1B_OFF);
  __bf16* w2b   = (__bf16*)(ws + W2B_OFF);
  float*  ybuf  = (float*)(ws + YBUF_OFF);
  int*    te    = (int*)(ws + TE_OFF);
  float*  tw    = (float*)(ws + TW_OFF);
  int*    rowTok = (int*)(ws + RT_OFF);
  int*    rowK   = (int*)(ws + RK_OFF);
  float*  rowW   = (float*)(ws + RW_OFF);
  int*    cnt   = (int*)(ws + CNT_OFF);
  int*    offs  = (int*)(ws + OFFS_OFF);
  int*    cur   = (int*)(ws + CUR_OFF);

  const int n4w = (NEXP * DFF * DMODEL) / 4;                 // 8,388,608 float4s
  k_init<<<1, 32, 0, stream>>>(cnt);
  k_cvt<<<(n4w + 255) / 256, 256, 0, stream>>>(w1, w1b, n4w);
  k_cvt<<<(n4w + 255) / 256, 256, 0, stream>>>(w2, w2b, n4w);
  k_gate<<<NTOK / 256, 256, 0, stream>>>(x, gate_w, te, tw, cnt);
  k_offs<<<1, 256, 0, stream>>>(cnt, offs, cur, rowTok);
  k_assign<<<NTOK / 256, 256, 0, stream>>>(te, tw, cur, rowTok, rowK, rowW);
  k_ffn<<<NTILES, 512, SMEM_BYTES, stream>>>(x, w1b, w2b, b1, b2,
                                             rowTok, rowK, rowW, offs, ybuf);
  k_combine<<<(NTOK * DMODEL / 4) / 256, 256, 0, stream>>>(ybuf, out);
}
